// FlashAttention_81432579932631
// MI455X (gfx1250) — matrix operand order repaired and checked by reading, not compiled, not run on hardware
//
#include <hip/hip_runtime.h>

typedef __attribute__((ext_vector_type(16))) _Float16 v16h;
typedef __attribute__((ext_vector_type(8)))  _Float16 v8h;
typedef __attribute__((ext_vector_type(4)))  _Float16 v4h;
typedef __attribute__((ext_vector_type(2)))  _Float16 v2h;
typedef __attribute__((ext_vector_type(8)))  float    v8f;

#define BATCH   4
#define HEADS   16
#define SEQ     4096
#define DH      64
#define QB      128         // Q rows per workgroup (8 waves x 16)
#define KB      64          // KV rows per iteration
#define NB      (SEQ / KB)
#define NWAVE   8
#define NTHR    (NWAVE * 32)
#define LSTRIDE 72          // padded LDS row stride in halves (144B: 16B aligned, bank-spread)
#define NEG_INF (-1e10f)
#define EPS     (1e-10f)

static __device__ __forceinline__ v16h lds_v16h(const _Float16* p) {
  v8h lo = *(const v8h*)(p);
  v8h hi = *(const v8h*)(p + 8);
  return __builtin_shufflevector(lo, hi, 0,1,2,3,4,5,6,7,8,9,10,11,12,13,14,15);
}

// packed f16 xor-butterfly max: one ds_swizzle_b32 reduces two rows at once
// (group-of-32 mode: offset[14:10]=xor_mask, [4:0]=and_mask)
template <int IMM>
static __device__ __forceinline__ v2h swz_max_h2(v2h v) {
  int oi = __builtin_amdgcn_ds_swizzle(__builtin_bit_cast(int, v), IMM);
  v2h o = __builtin_bit_cast(v2h, oi);
  v2h r;
  r[0] = (v[0] > o[0]) ? v[0] : o[0];
  r[1] = (v[1] > o[1]) ? v[1] : o[1];
  return r;
}

__global__ __launch_bounds__(NTHR)
void fa_fwd_f16wmma(const float* __restrict__ Q, const float* __restrict__ K,
                    const float* __restrict__ V, float* __restrict__ O)
{
  __shared__ alignas(32) _Float16 Ks [KB * LSTRIDE];            // K block, [kv][d]
  __shared__ alignas(32) _Float16 Vts[DH * LSTRIDE];            // V block transposed, [d][kv]
  __shared__ alignas(32) _Float16 Ps [NWAVE * 16 * LSTRIDE];    // per-wave P strip, [m][kv]

  const int tid  = threadIdx.x;
  const int lane = tid & 31;
  const int wave = tid >> 5;
  const int n16  = lane & 15;        // N (and A-row M) index within 16-lane half
  const int hi   = lane >> 4;        // which 16-lane half
  const int klo  = hi ? 8 : 0;       // A-fragment K sub-offset (16-bit A layout)
  const int kbs  = hi ? 16 : 0;      // B-fragment K base       (contiguous k = 16h + s; A uses it too)

  const int qblk = blockIdx.x;                 // 0..31
  const int bh   = blockIdx.y;                 // 0..63
  const size_t head = (size_t)bh * SEQ * DH;
  const float* Qg = Q + head;
  const float* Kg = K + head;
  const float* Vg = V + head;
  float*       Og = O + head;

  // all-ones B matrix: D = P x 1 gives row-sums of P replicated across all N lanes
  v16h bOnes;
  #pragma unroll
  for (int i = 0; i < 16; ++i) bOnes[i] = (_Float16)1.0f;

  // ---- Q A-fragments (16x32 f16 each); fold softmax scale AND log2(e) into Q
  const float scale = 0.125f * 1.44269504088896340736f;
  const int   qrow  = qblk * QB + wave * 16 + n16;   // A-matrix: M = lane&15
  v16h aQ[2];
  {
    const float* qp = Qg + (size_t)qrow * DH;
    #pragma unroll
    for (int f = 0; f < 2; ++f) {
      #pragma unroll
      for (int c = 0; c < 2; ++c) {              // halves 8c..8c+7 <- d = 32f + kbs + 8c + 0..7
        const float* p = qp + 32 * f + 8 * c + kbs;
        float4 x0 = *(const float4*)(p);
        float4 x1 = *(const float4*)(p + 4);
        int b = 8 * c;
        aQ[f][b+0] = (_Float16)(x0.x * scale); aQ[f][b+1] = (_Float16)(x0.y * scale);
        aQ[f][b+2] = (_Float16)(x0.z * scale); aQ[f][b+3] = (_Float16)(x0.w * scale);
        aQ[f][b+4] = (_Float16)(x1.x * scale); aQ[f][b+5] = (_Float16)(x1.y * scale);
        aQ[f][b+6] = (_Float16)(x1.z * scale); aQ[f][b+7] = (_Float16)(x1.w * scale);
      }
    }
  }

  // ---- register-pipelined staging: 2-row x 4-col tiles, loads for block kb+1
  //      are issued before the compute phase and stored to LDS next iteration.
  float4 skx[2][2], svx[2][2];   // [it][row-in-pair]
  auto stage_load = [&](int kb) {
    const float* kp = Kg + (size_t)kb * KB * DH;
    const float* vp = Vg + (size_t)kb * KB * DH;
    #pragma unroll
    for (int it = 0; it < 2; ++it) {
      int t  = tid + NTHR * it;        // 0..511 tiles
      int R  = t & 31;                 // row pair -> rows 2R, 2R+1
      int c4 = (t >> 5) << 2;          // col group 0..60
      const float* k0 = kp + (2 * R) * DH + c4;
      const float* v0 = vp + (2 * R) * DH + c4;
      skx[it][0] = *(const float4*)(k0);
      skx[it][1] = *(const float4*)(k0 + DH);
      svx[it][0] = *(const float4*)(v0);
      svx[it][1] = *(const float4*)(v0 + DH);
    }
  };
  auto stage_store = [&]() {
    #pragma unroll
    for (int it = 0; it < 2; ++it) {
      int t  = tid + NTHR * it;
      int R  = t & 31;
      int c4 = (t >> 5) << 2;
      #pragma unroll
      for (int rr = 0; rr < 2; ++rr) {           // K: row-major v4h stores
        float4 kx = skx[it][rr];
        v4h kh;
        kh[0]=(_Float16)kx.x; kh[1]=(_Float16)kx.y;
        kh[2]=(_Float16)kx.z; kh[3]=(_Float16)kx.w;
        *(v4h*)&Ks[(2 * R + rr) * LSTRIDE + c4] = kh;
      }
      float4 va = svx[it][0], vb = svx[it][1];   // V: transposed, packed-pair b32 stores
      v2h p;
      p[0]=(_Float16)va.x; p[1]=(_Float16)vb.x; *(v2h*)&Vts[(c4+0)*LSTRIDE + 2*R] = p;
      p[0]=(_Float16)va.y; p[1]=(_Float16)vb.y; *(v2h*)&Vts[(c4+1)*LSTRIDE + 2*R] = p;
      p[0]=(_Float16)va.z; p[1]=(_Float16)vb.z; *(v2h*)&Vts[(c4+2)*LSTRIDE + 2*R] = p;
      p[0]=(_Float16)va.w; p[1]=(_Float16)vb.w; *(v2h*)&Vts[(c4+3)*LSTRIDE + 2*R] = p;
    }
  };

  // ---- accumulators: U = unnormalized O (4 d-tiles), running row max / row sum (base-2)
  v8f U[4];
  float m_run[8], l_run[8];
  #pragma unroll
  for (int j = 0; j < 4; ++j) { v8f z{}; U[j] = z; }
  #pragma unroll
  for (int r = 0; r < 8; ++r) { m_run[r] = NEG_INF; l_run[r] = 0.0f; }

  _Float16* pw = &Ps[wave * 16 * LSTRIDE];

  stage_load(0);
  for (int kb = 0; kb < NB; ++kb) {
    __syncthreads();   // previous iteration's LDS reads complete before restage
    stage_store();
    __syncthreads();
    if (kb + 2 < NB) {   // HBM->L2 leg, two blocks ahead
      __builtin_prefetch(Kg + (size_t)(kb + 2) * KB * DH + tid * 16, 0, 3);
      __builtin_prefetch(Vg + (size_t)(kb + 2) * KB * DH + tid * 16, 0, 3);
    }
    if (kb + 1 < NB) stage_load(kb + 1);   // L2->VGPR leg hidden behind compute

    // ---- S = (Q*scale) . K^T : batch B-fragment loads, then 8 WMMAs ----
    v16h bk0[4], bk1[4];
    #pragma unroll
    for (int j = 0; j < 4; ++j) {
      const _Float16* kr = &Ks[(j * 16 + n16) * LSTRIDE + kbs];
      bk0[j] = lds_v16h(kr);          // K(d)=0..31
      bk1[j] = lds_v16h(kr + 32);     // K(d)=32..63
    }
    v8f Sv[4];
    #pragma unroll
    for (int j = 0; j < 4; ++j) {
      v8f c{};
      Sv[j] = __builtin_amdgcn_wmma_f32_16x16x32_f16(false, aQ[0], false, bk0[j], (short)0, c, false, false);
    }
    #pragma unroll
    for (int j = 0; j < 4; ++j)
      Sv[j] = __builtin_amdgcn_wmma_f32_16x16x32_f16(false, aQ[1], false, bk1[j], (short)0, Sv[j], false, false);

    // ---- row max: tile-reduce in VALU, then packed-f16 butterfly (2 rows per swizzle)
    float pm[8];
    #pragma unroll
    for (int r = 0; r < 8; ++r)
      pm[r] = fmaxf(fmaxf(Sv[0][r], Sv[1][r]), fmaxf(Sv[2][r], Sv[3][r]));
    v2h mpk[4];
    #pragma unroll
    for (int i = 0; i < 4; ++i)
      mpk[i] = __builtin_bit_cast(v2h, __builtin_amdgcn_cvt_pkrtz(pm[2*i], pm[2*i+1]));
    #pragma unroll
    for (int i = 0; i < 4; ++i) {
      v2h v = mpk[i];
      v = swz_max_h2<0x041F>(v);   // xor 1
      v = swz_max_h2<0x081F>(v);   // xor 2
      v = swz_max_h2<0x101F>(v);   // xor 4
      v = swz_max_h2<0x201F>(v);   // xor 8
      mpk[i] = v;
    }
    float alpha[8];
    #pragma unroll
    for (int i = 0; i < 4; ++i) {
      float lo  = (float)mpk[i][0];
      float hiv = (float)mpk[i][1];
      float mn0 = fmaxf(m_run[2*i],   lo);
      float mn1 = fmaxf(m_run[2*i+1], hiv);
      alpha[2*i]   = __builtin_amdgcn_exp2f(m_run[2*i]   - mn0);   // raw v_exp_f32
      alpha[2*i+1] = __builtin_amdgcn_exp2f(m_run[2*i+1] - mn1);
      m_run[2*i]   = mn0;
      m_run[2*i+1] = mn1;
    }

    // ---- P = exp2(S - m), straight to LDS as f16 (row-sum comes from the ones-WMMA) ----
    #pragma unroll
    for (int j = 0; j < 4; ++j)
      #pragma unroll
      for (int r = 0; r < 8; ++r) {
        float p = __builtin_amdgcn_exp2f(Sv[j][r] - m_run[r]);
        int m = r + (hi << 3);
        pw[m * LSTRIDE + j * 16 + n16] = (_Float16)p;
      }
    #pragma unroll
    for (int j = 0; j < 4; ++j)
      #pragma unroll
      for (int r = 0; r < 8; ++r) U[j][r] *= alpha[r];
    __builtin_amdgcn_wave_barrier();   // wave-private LDS: same-wave DS ops are in order

    v16h aP[2];
    {
      const _Float16* pr = pw + n16 * LSTRIDE;   // A: M = lane&15
      #pragma unroll
      for (int f = 0; f < 2; ++f) {
        v8h c0 = *(const v8h*)(pr + f * 32 + kbs);        // K(kv) = 32f+kbs+0..7
        v8h c1 = *(const v8h*)(pr + f * 32 + kbs + 8);    // K(kv) = 32f+kbs+8..15
        aP[f] = __builtin_shufflevector(c0, c1, 0,1,2,3,4,5,6,7,8,9,10,11,12,13,14,15);
      }
    }

    // ---- lsum = P . ones via the matrix pipe (C-layout, replicated across lanes) ----
    v8f Lv;
    {
      v8f z{};
      Lv = __builtin_amdgcn_wmma_f32_16x16x32_f16(false, aP[0], false, bOnes, (short)0, z,  false, false);
      Lv = __builtin_amdgcn_wmma_f32_16x16x32_f16(false, aP[1], false, bOnes, (short)0, Lv, false, false);
    }

    // ---- U += P . V : batch all 8 B-fragment loads, then 8 WMMAs ----
    v16h bv0[4], bv1[4];
    #pragma unroll
    for (int j = 0; j < 4; ++j) {
      const _Float16* vr = &Vts[(j * 16 + n16) * LSTRIDE + kbs];
      bv0[j] = lds_v16h(vr);          // K(kv)=0..31
      bv1[j] = lds_v16h(vr + 32);     // K(kv)=32..63
    }
    #pragma unroll
    for (int j = 0; j < 4; ++j)
      U[j] = __builtin_amdgcn_wmma_f32_16x16x32_f16(false, aP[0], false, bv0[j], (short)0, U[j], false, false);
    #pragma unroll
    for (int j = 0; j < 4; ++j)
      U[j] = __builtin_amdgcn_wmma_f32_16x16x32_f16(false, aP[1], false, bv1[j], (short)0, U[j], false, false);

    #pragma unroll
    for (int r = 0; r < 8; ++r)
      l_run[r] = alpha[r] * l_run[r] + Lv[r] + EPS;
  }

  // ---- epilogue: O = U * (1/l) ----
  float linv[8];
  #pragma unroll
  for (int r = 0; r < 8; ++r) linv[r] = __builtin_amdgcn_rcpf(l_run[r]);
  #pragma unroll
  for (int j = 0; j < 4; ++j)
    #pragma unroll
    for (int r = 0; r < 8; ++r) {
      int m   = r + (hi << 3);
      int row = qblk * QB + wave * 16 + m;
      Og[(size_t)row * DH + j * 16 + n16] = U[j][r] * linv[r];
    }
}

extern "C" void kernel_launch(void* const* d_in, const int* in_sizes, int n_in,
                              void* d_out, int out_size, void* d_ws, size_t ws_size,
                              hipStream_t stream) {
  const float* Q = (const float*)d_in[0];
  const float* K = (const float*)d_in[1];
  const float* V = (const float*)d_in[2];
  float* O = (float*)d_out;
  dim3 grid(SEQ / QB, BATCH * HEADS);
  fa_fwd_f16wmma<<<grid, NTHR, 0, stream>>>(Q, K, V, O);
}
